// CosineModel_28003186770375
// MI455X (gfx1250) — matrix operand order repaired and checked by reading, not compiled, not run on hardware
//
#include <hip/hip_runtime.h>
#include <hip/hip_bf16.h>
#include <cstdint>

// ---------------------------------------------------------------------------
// CDNA5 (gfx1250) wave32 WMMA implementation of the CosineModel pipeline.
//   - 48 fused LSTM-step kernels:  gates = [embed(x_s) | h_prev] x [Wih|Whh]^T
//     via v_wmma_f32_16x16x32_bf16; weight tiles staged global->LDS with the
//     gfx1250 async-to-LDS path (ASYNCcnt), double-buffered; LSTM cell fused.
//   - MLP head: BN stats -> BN apply (bf16 cvt) -> WMMA GEMM (+bias+ReLU).
// ---------------------------------------------------------------------------

typedef __attribute__((ext_vector_type(16))) __bf16 v16bf;
typedef __attribute__((ext_vector_type(8)))  __bf16 v8bf;
typedef __attribute__((ext_vector_type(8)))  float  v8f;
typedef __attribute__((vector_size(16)))     int    v4i_;   // builtin's b128 unit

#define SEQ      48
#define BATCH    4096
#define BATCH2   8192          // premise + hypothesis fused
#define D_EMB    300
#define KX       320           // D_EMB padded to mult of 32
#define D_HID    512
#define GATES    2048          // 4*H
#define SEQ_IN   1025          // 2H + 1
#define KP_MLP   1056          // SEQ_IN padded to mult of 32
#define NP_MLP   1152          // SEQ_IN padded to mult of 128 (tile N)
#define D_OUT    3
#define BN_EPS   1e-5f

#define LDS_STRIDE 40          // 32 k-elems + 8 pad (80B) -> conflict-free b128 DS reads

// ---------------- gfx1250 async global->LDS helpers ----------------

#if defined(__has_builtin)
#if __has_builtin(__builtin_amdgcn_global_load_async_to_lds_b128) && \
    __has_builtin(__builtin_amdgcn_s_wait_asynccnt)
#define HAVE_ASYNC_LDS 1
#endif
#endif
#ifndef HAVE_ASYNC_LDS
#define HAVE_ASYNC_LDS 0
#endif

// copy 16 bytes (8 bf16) global -> LDS
__device__ __forceinline__ void cp16_g2l(__bf16* lds, const __bf16* g) {
#if HAVE_ASYNC_LDS
  __builtin_amdgcn_global_load_async_to_lds_b128((v4i_*)g, (v4i_*)lds, 0, 0);
#else
  *(v8bf*)lds = *(const v8bf*)g;
#endif
}
__device__ __forceinline__ void async_wait_le2() {
#if HAVE_ASYNC_LDS
  __builtin_amdgcn_s_wait_asynccnt(2);
#endif
}
__device__ __forceinline__ void async_wait_0() {
#if HAVE_ASYNC_LDS
  __builtin_amdgcn_s_wait_asynccnt(0);
#endif
}

// ---------------- misc helpers ----------------

__device__ __forceinline__ __bf16 f2bf(float f) {
  union { float f; unsigned u; } a; a.f = f;
  unsigned u = a.u + (0x7FFFu + ((a.u >> 16) & 1u));   // round-to-nearest-even
  union { unsigned short s; __bf16 b; } r; r.s = (unsigned short)(u >> 16);
  return r.b;
}

__device__ __forceinline__ float sigmoidf_(float x) {
  return 1.0f / (1.0f + __expf(-x));
}

// A fragment (16x32 bf16): lane<16 -> row=lane, K=0..15;
// lane>=16 -> row=lane-16, K=16..31.  rowPtr = &A[thisLaneRow][0].
__device__ __forceinline__ v16bf load_a_frag(const __bf16* rowPtr, int k) {
  const int lane = threadIdx.x & 31;
  const int kb = k + ((lane & 16) ? 16 : 0);
  union { v16bf v; v8bf h[2]; } u;
  u.h[0] = *(const v8bf*)(rowPtr + kb);
  u.h[1] = *(const v8bf*)(rowPtr + kb + 8);
  return u.v;
}

// B fragment (32x16 bf16): lane<16 -> col=lane, K=0..15; lane>=16 -> col=lane-16,
// K=16..31. With B[k][n] = W[n][k] (row-major weights), K is contiguous.
__device__ __forceinline__ v16bf load_b_frag(const __bf16* colPtr, int k) {
  const int lane = threadIdx.x & 31;
  const int kb = k + ((lane & 16) ? 16 : 0);
  union { v16bf v; v8bf h[2]; } u;
  u.h[0] = *(const v8bf*)(colPtr + kb);
  u.h[1] = *(const v8bf*)(colPtr + kb + 8);
  return u.v;
}

// B fragment from an LDS tile laid out as [128 cols][LDS_STRIDE bf16]
__device__ __forceinline__ v16bf lds_b_frag(const __bf16* tile, int f) {
  const int lane = threadIdx.x & 31;
  const int ln = lane & 15;
  const int kb = (lane & 16) ? 16 : 0;
  const __bf16* p = tile + (f * 16 + ln) * LDS_STRIDE + kb;
  union { v16bf v; v8bf h[2]; } u;
  u.h[0] = *(const v8bf*)p;
  u.h[1] = *(const v8bf*)(p + 8);
  return u.v;
}

#define WMMA_BF16(acc, a, b) \
  acc = __builtin_amdgcn_wmma_f32_16x16x32_bf16(false, a, false, b, (short)0, acc, false, false)

// ---------------- utility kernels ----------------

__global__ void zero_u32(unsigned* p, long n) {
  long i = (long)blockIdx.x * 256 + threadIdx.x;
  if (i < n) p[i] = 0u;
}

__global__ void cvt_pad_bf16(const float* __restrict__ src, __bf16* __restrict__ dst,
                             int rowsSrc, int rowsDst, int Ksrc, int Kp) {
  long i = (long)blockIdx.x * 256 + threadIdx.x;
  long n = (long)rowsDst * Kp;
  if (i >= n) return;
  int r = (int)(i / Kp);
  int k = (int)(i - (long)r * Kp);
  float v = (r < rowsSrc && k < Ksrc) ? src[(long)r * Ksrc + k] : 0.0f;
  dst[i] = f2bf(v);
}

__global__ void pad_f32(const float* __restrict__ src, float* __restrict__ dst,
                        int n, int np) {
  int i = blockIdx.x * 256 + threadIdx.x;
  if (i < np) dst[i] = (i < n) ? src[i] : 0.0f;
}

// ---------------- fused LSTM step ----------------
// grid = (BATCH2/128, D_HID/32); block = 256 (8 waves).
// Wave w owns rows [blk*128 + w*16, +16).  Block owns units [blockIdx.y*32, +32).
// Gate tile = 128 rows x 128 gate-cols ({i,f,g,o} x 32 units); the B tile
// (128 cols x 32 k, 8KB) is staged global->LDS per k-step (async, dbl-buffered).
__global__ __launch_bounds__(256) void lstm_step(
    const __bf16* __restrict__ embedBf,   // [VOCAB][KX]
    const __bf16* __restrict__ WihBf,     // [GATES][KX]
    const __bf16* __restrict__ WhhBf,     // [GATES][D_HID]
    const float*  __restrict__ b_ih,
    const float*  __restrict__ b_hh,
    const int*    __restrict__ prem,      // [SEQ][BATCH]
    const int*    __restrict__ hyp,
    int s,
    const __bf16* __restrict__ hPrev,     // [BATCH2][D_HID]
    __bf16*       __restrict__ hNext,
    float*        __restrict__ cbuf,      // [BATCH2][D_HID]
    float*        __restrict__ hFinal,    // [BATCH2][D_HID]
    int isLast)
{
  __shared__ __bf16 Bt[2][128 * LDS_STRIDE];   // 2 x 10KB weight tile buffers

  const int tid  = threadIdx.x;
  const int wave = tid >> 5;
  const int lane = tid & 31;
  const int ln   = lane & 15;
  const int rowBase = blockIdx.x * 128 + wave * 16;
  const int u0      = blockIdx.y * 32;

  // --- A row pointers for this lane ---
  const int myRow = rowBase + ln;
  const int b   = myRow & (BATCH - 1);
  const int sel = myRow >> 12;
  const int tok = sel ? hyp[s * BATCH + b] : prem[s * BATCH + b];
  const __bf16* aX = embedBf + (long)tok * KX;
  const __bf16* aH = hPrev + (long)myRow * D_HID;

  // --- cooperative B-tile copy: thread t -> col c = t/2, 32B half = t&1 ---
  const int c    = tid >> 1;                     // 0..127
  const int hOff = (tid & 1) * 16;               // element offset 0 / 16
  const int colG = (c >> 5) * D_HID + u0 + (c & 31);
  const __bf16* srcX = WihBf + (long)colG * KX    + hOff;
  const __bf16* srcH = WhhBf + (long)colG * D_HID + hOff;
  __bf16* dst0 = &Bt[0][0] + c * LDS_STRIDE + hOff;

  const int NKX = KX / 32;            // 10 k-steps over embedding
  const int NK  = NKX + D_HID / 32;   // + 16 k-steps over h  = 26

  v8f acc[8];
#pragma unroll
  for (int f = 0; f < 8; ++f)
#pragma unroll
    for (int v = 0; v < 8; ++v) acc[f][v] = 0.0f;

  // issue the B-tile copy for k-step 'st' into LDS buffer 'bf'
  auto issue = [&](int st, int bf) {
    const __bf16* src = (st < NKX) ? (srcX + st * 32) : (srcH + (st - NKX) * 32);
    __bf16* d = dst0 + bf * (128 * LDS_STRIDE);
    cp16_g2l(d, src);
    cp16_g2l(d + 8, src + 8);
  };

  issue(0, 0);
  for (int st = 0; st < NK; ++st) {
    const int bf = st & 1;
    if (st + 1 < NK) { issue(st + 1, bf ^ 1); async_wait_le2(); }
    else             { async_wait_0(); }
    __syncthreads();                              // tile 'bf' visible to all waves

    v16bf a = (st < NKX) ? load_a_frag(aX, st * 32)
                         : load_a_frag(aH, (st - NKX) * 32);
    const __bf16* tile = &Bt[bf][0];
#pragma unroll
    for (int f = 0; f < 8; ++f) {
      v16bf bb = lds_b_frag(tile, f);
      WMMA_BF16(acc[f], a, bb);
    }
    __syncthreads();                              // done reading before overwrite
  }

  // --- biases (depend only on ln; same for both lane halves) ---
  float bias[8];
#pragma unroll
  for (int g = 0; g < 4; ++g)
#pragma unroll
    for (int t = 0; t < 2; ++t) {
      int col = g * D_HID + u0 + t * 16 + ln;
      bias[g * 2 + t] = b_ih[col] + b_hh[col];
    }

  // --- LSTM cell epilogue ---
  const int rOff = (lane & 16) ? 8 : 0;
#pragma unroll
  for (int t = 0; t < 2; ++t) {
    const int col = u0 + t * 16 + ln;
#pragma unroll
    for (int v = 0; v < 8; ++v) {
      const int row = rowBase + v + rOff;
      float ig = sigmoidf_(acc[0 * 2 + t][v] + bias[0 * 2 + t]);
      float fg = sigmoidf_(acc[1 * 2 + t][v] + bias[1 * 2 + t]);
      float gg = tanhf    (acc[2 * 2 + t][v] + bias[2 * 2 + t]);
      float og = sigmoidf_(acc[3 * 2 + t][v] + bias[3 * 2 + t]);
      const long idx = (long)row * D_HID + col;
      float cc = fg * cbuf[idx] + ig * gg;
      cbuf[idx] = cc;
      float h = og * tanhf(cc);
      hNext[idx] = f2bf(h);
      if (isLast) hFinal[idx] = h;
    }
  }
}

// ---------------- cosine distance + feature build ----------------
__global__ __launch_bounds__(256) void cosine_feat(const float* __restrict__ hFinal,
                                                   float* __restrict__ feat)
{
  const int row  = blockIdx.x * 8 + (threadIdx.x >> 5);
  const int lane = threadIdx.x & 31;
  const float* p = hFinal + (long)row * D_HID;
  const float* h = hFinal + (long)(row + BATCH) * D_HID;
  float dp = 0.f, np = 0.f, nh = 0.f;
  for (int k = lane; k < D_HID; k += 32) {
    float a = p[k], c = h[k];
    dp += a * c; np += a * a; nh += c * c;
    feat[(long)row * SEQ_IN + 1 + k]         = a;
    feat[(long)row * SEQ_IN + 1 + D_HID + k] = c;
  }
#pragma unroll
  for (int m = 16; m >= 1; m >>= 1) {
    dp += __shfl_xor(dp, m, 32);
    np += __shfl_xor(np, m, 32);
    nh += __shfl_xor(nh, m, 32);
  }
  if (lane == 0)
    feat[(long)row * SEQ_IN] = 1.0f - dp / (sqrtf(np) * sqrtf(nh));
}

// ---------------- batchnorm stats (one block per column) ----------------
__global__ __launch_bounds__(256) void bn_stats(const float* __restrict__ X,
                                                int M, int ld,
                                                float* __restrict__ mean,
                                                float* __restrict__ rstd)
{
  const int col = blockIdx.x;
  const int tid = threadIdx.x;
  float s = 0.f, s2 = 0.f;
  for (int r = tid; r < M; r += 256) {
    float v = X[(long)r * ld + col];
    s += v; s2 += v * v;
  }
  __shared__ float sh[2][256];
  sh[0][tid] = s; sh[1][tid] = s2;
  __syncthreads();
  for (int step = 128; step > 0; step >>= 1) {
    if (tid < step) {
      sh[0][tid] += sh[0][tid + step];
      sh[1][tid] += sh[1][tid + step];
    }
    __syncthreads();
  }
  if (tid == 0) {
    float m = sh[0][0] / (float)M;
    float v = sh[1][0] / (float)M - m * m;
    mean[col] = m;
    rstd[col] = rsqrtf(v + BN_EPS);
  }
}

__global__ void bn_apply_bf16(const float* __restrict__ X, int M, int N, int ld, int Kp,
                              const float* __restrict__ mean, const float* __restrict__ rstd,
                              const float* __restrict__ gamma, const float* __restrict__ beta,
                              __bf16* __restrict__ Y)
{
  long i = (long)blockIdx.x * 256 + threadIdx.x;
  if (i >= (long)M * Kp) return;
  int m = (int)(i / Kp);
  int k = (int)(i - (long)m * Kp);
  float v = 0.0f;
  if (k < N) v = (X[(long)m * ld + k] - mean[k]) * rstd[k] * gamma[k] + beta[k];
  Y[i] = f2bf(v);
}

__global__ void bn_apply_f32(const float* __restrict__ X, int M, int N, int ld,
                             const float* __restrict__ mean, const float* __restrict__ rstd,
                             const float* __restrict__ gamma, const float* __restrict__ beta,
                             float* __restrict__ Y)
{
  long i = (long)blockIdx.x * 256 + threadIdx.x;
  if (i >= (long)M * N) return;
  int m = (int)(i / N);
  int k = (int)(i - (long)m * N);
  Y[i] = (X[(long)m * ld + k] - mean[k]) * rstd[k] * gamma[k] + beta[k];
}

// ---------------- WMMA GEMM: C[M,Np] = A[M,Kp] * B[Np,Kp]^T + bias (+ReLU) ----
__global__ __launch_bounds__(256) void gemm_bf16(
    const __bf16* __restrict__ A, const __bf16* __restrict__ B,
    float* __restrict__ C, const float* __restrict__ bias,
    int Kp, int ldc, int relu)
{
  const int tid  = threadIdx.x;
  const int wave = tid >> 5;
  const int lane = tid & 31;
  const int ln   = lane & 15;
  const int rowBase = blockIdx.x * 128 + wave * 16;
  const int colBase = blockIdx.y * 128;

  const __bf16* aRow = A + (long)(rowBase + ln) * Kp;
  const __bf16* bCol[8];
#pragma unroll
  for (int f = 0; f < 8; ++f)
    bCol[f] = B + (long)(colBase + f * 16 + ln) * Kp;

  v8f acc[8];
#pragma unroll
  for (int f = 0; f < 8; ++f)
#pragma unroll
    for (int v = 0; v < 8; ++v) acc[f][v] = 0.0f;

  for (int k = 0; k < Kp; k += 32) {
    v16bf a = load_a_frag(aRow, k);
#pragma unroll
    for (int f = 0; f < 8; ++f) {
      v16bf bb = load_b_frag(bCol[f], k);
      WMMA_BF16(acc[f], a, bb);
    }
  }

  const int rOff = (lane & 16) ? 8 : 0;
#pragma unroll
  for (int f = 0; f < 8; ++f) {
    const int col = colBase + f * 16 + ln;
    const float bs = bias[col];
#pragma unroll
    for (int v = 0; v < 8; ++v) {
      float r = acc[f][v] + bs;
      if (relu) r = fmaxf(r, 0.0f);
      C[(long)(rowBase + v + rOff) * ldc + col] = r;
    }
  }
}

// ---------------- final tiny linear [B,1025] x [3,1025]^T ----------------
__global__ void final_linear(const float* __restrict__ xnf,
                             const float* __restrict__ Wo,
                             const float* __restrict__ blo,
                             float* __restrict__ out)
{
  int i = blockIdx.x * 256 + threadIdx.x;
  if (i >= BATCH * D_OUT) return;
  int b = i / D_OUT, j = i - b * D_OUT;
  const float* x = xnf + (long)b * SEQ_IN;
  const float* w = Wo  + (long)j * SEQ_IN;
  float s = blo[j];
  for (int k = 0; k < SEQ_IN; ++k) s += x[k] * w[k];
  out[i] = s;
}

// ---------------------------------------------------------------------------

extern "C" void kernel_launch(void* const* d_in, const int* in_sizes, int n_in,
                              void* d_out, int out_size, void* d_ws, size_t ws_size,
                              hipStream_t stream) {
  const int*   prem  = (const int*)  d_in[0];
  const int*   hyp   = (const int*)  d_in[1];
  const float* embed = (const float*)d_in[2];
  const float* W_ih  = (const float*)d_in[3];
  const float* W_hh  = (const float*)d_in[4];
  const float* b_ih  = (const float*)d_in[5];
  const float* b_hh  = (const float*)d_in[6];
  const float* g0    = (const float*)d_in[7];
  const float* be0   = (const float*)d_in[8];
  const float* W0    = (const float*)d_in[9];
  const float* bl0   = (const float*)d_in[10];
  const float* g1    = (const float*)d_in[11];
  const float* be1   = (const float*)d_in[12];
  const float* W1    = (const float*)d_in[13];
  const float* bl1   = (const float*)d_in[14];
  const float* go    = (const float*)d_in[15];
  const float* beo   = (const float*)d_in[16];
  const float* Wo    = (const float*)d_in[17];
  const float* blo   = (const float*)d_in[18];
  float* out = (float*)d_out;

  const int VOCAB = 50000;

  // ---- carve workspace ----
  char* w = (char*)d_ws;
  size_t off = 0;
  auto carve = [&](size_t bytes) -> void* {
    void* p = w + off;
    off += (bytes + 255) & ~(size_t)255;
    return p;
  };
  __bf16* embedBf = (__bf16*)carve((size_t)VOCAB * KX * 2);        // 32 MB
  __bf16* WihBf   = (__bf16*)carve((size_t)GATES * KX * 2);
  __bf16* WhhBf   = (__bf16*)carve((size_t)GATES * D_HID * 2);
  __bf16* hA      = (__bf16*)carve((size_t)BATCH2 * D_HID * 2);
  __bf16* hB      = (__bf16*)carve((size_t)BATCH2 * D_HID * 2);
  float*  cbuf    = (float*) carve((size_t)BATCH2 * D_HID * 4);
  float*  hFinal  = (float*) carve((size_t)BATCH2 * D_HID * 4);
  float*  feat    = (float*) carve((size_t)BATCH * SEQ_IN * 4);
  float*  meanb   = (float*) carve((size_t)NP_MLP * 4);
  float*  rstdb   = (float*) carve((size_t)NP_MLP * 4);
  __bf16* xbf     = (__bf16*)carve((size_t)BATCH * KP_MLP * 2);
  __bf16* W0bf    = (__bf16*)carve((size_t)NP_MLP * KP_MLP * 2);
  __bf16* W1bf    = (__bf16*)carve((size_t)NP_MLP * KP_MLP * 2);
  float*  b0p     = (float*) carve((size_t)NP_MLP * 4);
  float*  b1p     = (float*) carve((size_t)NP_MLP * 4);
  float*  y1      = (float*) carve((size_t)BATCH * NP_MLP * 4);
  float*  y2      = (float*) carve((size_t)BATCH * NP_MLP * 4);
  float*  xnf     = (float*) carve((size_t)BATCH * SEQ_IN * 4);
  (void)ws_size; (void)in_sizes; (void)n_in; (void)out_size;

  auto blocks = [](long n) { return (unsigned)((n + 255) / 256); };

  // ---- weight / embedding conversion (bf16, zero-padded) ----
  cvt_pad_bf16<<<blocks((long)VOCAB * KX), 256, 0, stream>>>(embed, embedBf, VOCAB, VOCAB, D_EMB, KX);
  cvt_pad_bf16<<<blocks((long)GATES * KX), 256, 0, stream>>>(W_ih, WihBf, GATES, GATES, D_EMB, KX);
  cvt_pad_bf16<<<blocks((long)GATES * D_HID), 256, 0, stream>>>(W_hh, WhhBf, GATES, GATES, D_HID, D_HID);
  cvt_pad_bf16<<<blocks((long)NP_MLP * KP_MLP), 256, 0, stream>>>(W0, W0bf, SEQ_IN, NP_MLP, SEQ_IN, KP_MLP);
  cvt_pad_bf16<<<blocks((long)NP_MLP * KP_MLP), 256, 0, stream>>>(W1, W1bf, SEQ_IN, NP_MLP, SEQ_IN, KP_MLP);
  pad_f32<<<blocks(NP_MLP), 256, 0, stream>>>(bl0, b0p, SEQ_IN, NP_MLP);
  pad_f32<<<blocks(NP_MLP), 256, 0, stream>>>(bl1, b1p, SEQ_IN, NP_MLP);

  // ---- zero initial state (h0, c0) ----
  zero_u32<<<blocks((long)BATCH2 * D_HID / 2), 256, 0, stream>>>((unsigned*)hA, (long)BATCH2 * D_HID / 2);
  zero_u32<<<blocks((long)BATCH2 * D_HID), 256, 0, stream>>>((unsigned*)cbuf, (long)BATCH2 * D_HID);

  // ---- recurrence: 48 fused WMMA step kernels ----
  dim3 sgrid(BATCH2 / 128, D_HID / 32);
  for (int s = 0; s < SEQ; ++s) {
    const __bf16* hp = (s & 1) ? hB : hA;
    __bf16*       hn = (s & 1) ? hA : hB;
    lstm_step<<<sgrid, 256, 0, stream>>>(embedBf, WihBf, WhhBf, b_ih, b_hh,
                                         prem, hyp, s, hp, hn, cbuf, hFinal,
                                         (s == SEQ - 1) ? 1 : 0);
  }

  // ---- cosine distance + feature assembly ----
  cosine_feat<<<BATCH / 8, 256, 0, stream>>>(hFinal, feat);

  // ---- MLP layer 0 ----
  bn_stats<<<SEQ_IN, 256, 0, stream>>>(feat, BATCH, SEQ_IN, meanb, rstdb);
  bn_apply_bf16<<<blocks((long)BATCH * KP_MLP), 256, 0, stream>>>(
      feat, BATCH, SEQ_IN, SEQ_IN, KP_MLP, meanb, rstdb, g0, be0, xbf);
  gemm_bf16<<<dim3(BATCH / 128, NP_MLP / 128), 256, 0, stream>>>(
      xbf, W0bf, y1, b0p, KP_MLP, NP_MLP, 1);

  // ---- MLP layer 1 ----
  bn_stats<<<SEQ_IN, 256, 0, stream>>>(y1, BATCH, NP_MLP, meanb, rstdb);
  bn_apply_bf16<<<blocks((long)BATCH * KP_MLP), 256, 0, stream>>>(
      y1, BATCH, SEQ_IN, NP_MLP, KP_MLP, meanb, rstdb, g1, be1, xbf);
  gemm_bf16<<<dim3(BATCH / 128, NP_MLP / 128), 256, 0, stream>>>(
      xbf, W1bf, y2, b1p, KP_MLP, NP_MLP, 1);

  // ---- output BN + final linear ----
  bn_stats<<<SEQ_IN, 256, 0, stream>>>(y2, BATCH, NP_MLP, meanb, rstdb);
  bn_apply_f32<<<blocks((long)BATCH * SEQ_IN), 256, 0, stream>>>(
      y2, BATCH, SEQ_IN, NP_MLP, meanb, rstdb, go, beo, xnf);
  final_linear<<<blocks(BATCH * D_OUT), 256, 0, stream>>>(xnf, Wo, blo, out);
}
